// Node2Property2_8701603742452
// MI455X (gfx1250) — matrix operand order repaired and checked by reading, not compiled, not run on hardware
//
#include <hip/hip_runtime.h>

// ---------------------------------------------------------------------------
// SchNet-style Node2Property: out = segment_sum( ssp(x@W1 + b1) @ W2, batch )
// N=1M, IN=H=128, OUT=1, G=16384.  HBM-bound (512MB of x @ 23.3TB/s ~ 22us).
// v3: same fragment-major WMMA pipeline as v2 (A via LDS b128, B via
// L2-resident pre-packed d_ws b128), plus a lean epilogue:
//   ssp(v) = ln2*log2(1 + exp2(v*log2e)) - ln2   (v_exp_f32 + v_log_f32)
// instead of libm log1pf/expf, and v_perm_b32-based bf16 packing.
// Requires ws_size >= 32KB.
// ---------------------------------------------------------------------------

typedef __attribute__((ext_vector_type(16))) __bf16       v16bf;
typedef __attribute__((ext_vector_type(8)))  float        v8f;
typedef __attribute__((ext_vector_type(8)))  unsigned int v8u;

constexpr int N_NODES        = 1 << 20;   // 1048576
constexpr int IN_DIM         = 128;
constexpr int H_DIM          = 128;
constexpr int N_GRAPHS       = 16384;
constexpr int ROWS_PER_BLOCK = 128;       // 8 waves x 16 rows
constexpr int THREADS        = 256;
constexpr float LN2F   = 0.69314718055994530942f;
constexpr float LOG2EF = 1.44269504088896340736f;

// Pack two fp32 into a bf16x2 dword with round-to-nearest-even.
// v_bfe+v_add per half, then one v_perm_b32 to merge the high halves.
__device__ __forceinline__ unsigned int pack_bf16x2(float lo, float hi) {
  unsigned int ulo = __builtin_bit_cast(unsigned int, lo);
  unsigned int uhi = __builtin_bit_cast(unsigned int, hi);
  ulo += 0x7FFFu + ((ulo >> 16) & 1u);
  uhi += 0x7FFFu + ((uhi >> 16) & 1u);
  // bytes: [0]=ulo.b2 [1]=ulo.b3 [2]=uhi.b2 [3]=uhi.b3
  return __builtin_amdgcn_perm(uhi, ulo, 0x07060302u);
}

// shifted softplus: softplus(v) - ln2, on raw v_exp_f32 / v_log_f32
__device__ __forceinline__ float ssp_fast(float v) {
  float e = __builtin_amdgcn_exp2f(v * LOG2EF);        // exp(v)
  float l = __builtin_amdgcn_logf(1.0f + e);           // log2(1+exp(v))
  return fmaf(LN2F, l, -LN2F);                         // ln2*log2(.) - ln2
}

__global__ void zero_out_kernel(float* __restrict__ out) {
  int i = blockIdx.x * blockDim.x + threadIdx.x;
  if (i < N_GRAPHS) out[i] = 0.0f;
}

// Pre-pack W1 (fp32 [128][128]) into B-fragment-major bf16 pairs in d_ws.
// Element g = ((kb*8 + nt)*32 + lane)*8 + v holds the packed K-pair
// p = 16*kb + 8*hi + v at column n = nt*16 + ln  (ln=lane&15, hi=lane>>4).
__global__ void pack_w1_kernel(const float* __restrict__ W1,
                               unsigned int* __restrict__ wsB) {
  int g    = blockIdx.x * blockDim.x + threadIdx.x;   // 0..8191
  int v    = g & 7;
  int lane = (g >> 3) & 31;
  int nt   = (g >> 8) & 7;
  int kb   = (g >> 11) & 3;
  int ln   = lane & 15;
  int hi   = lane >> 4;
  int p    = 16 * kb + 8 * hi + v;
  int n    = nt * 16 + ln;
  wsB[g] = pack_bf16x2(W1[(2 * p + 0) * H_DIM + n],
                       W1[(2 * p + 1) * H_DIM + n]);
}

__global__ __launch_bounds__(THREADS)
void mlp_pool_kernel(const float* __restrict__ x,
                     const long long* __restrict__ batch,
                     const float* __restrict__ b1,
                     const float* __restrict__ W2,
                     const v8u* __restrict__ wsBf,   // packed W1 fragments
                     float* __restrict__ out) {
  // A fragments, fragment-major: [wave][kb][lane] -> 8 contiguous dwords
  __shared__ v8u  sXf[8 * 4 * 32];          // 32 KB
  __shared__ float sB1[H_DIM];
  __shared__ float sW2[H_DIM];
  __shared__ float sRow[8][16];             // per-wave row scalars

  const int t        = threadIdx.x;
  const int gRowBase = blockIdx.x * ROWS_PER_BLOCK;
  unsigned int* sXd  = (unsigned int*)sXf;

  // ---- Stage x tile directly into A-fragment layout ----
  // For row, 4-pair group q (pairs kp=4q..4q+3): kb=q>>2, q'=q&3, hi=q'>>1,
  // vbase=4*(q'&1), lane=hi*16+(row&15); the 4 packed dwords land in
  // contiguous VGPR slots vbase..vbase+3 of that fragment (one b128 store).
  const float* xblk = x + (size_t)gRowBase * IN_DIM;
#pragma unroll
  for (int i = 0; i < 8; ++i) {
    int g4  = i * THREADS + t;      // 0..2047
    int row = g4 >> 4;              // 0..127
    int q   = g4 & 15;              // 4-pair group within row
    const float4* px = (const float4*)(xblk + row * IN_DIM + q * 8);
    float4 f0 = px[0];
    float4 f1 = px[1];
    uint4 d;
    d.x = pack_bf16x2(f0.x, f0.y);
    d.y = pack_bf16x2(f0.z, f0.w);
    d.z = pack_bf16x2(f1.x, f1.y);
    d.w = pack_bf16x2(f1.z, f1.w);
    int w     = row >> 4;
    int m     = row & 15;
    int kb    = q >> 2;
    int qq    = q & 3;
    int lane_ = (qq >> 1) * 16 + m;
    int vbase = 4 * (qq & 1);
    int addr  = ((w * 4 + kb) * 32 + lane_) * 8 + vbase;
    *(uint4*)(sXd + addr) = d;      // ds_store_b128
  }
  if (t < H_DIM) { sB1[t] = b1[t]; sW2[t] = W2[t]; }
  __syncthreads();

  const int lane = t & 31;
  const int wave = t >> 5;
  const int hi   = lane >> 4;
  const int ln   = lane & 15;

  v8f acc[8];
  const v8f vzero = {0.f, 0.f, 0.f, 0.f, 0.f, 0.f, 0.f, 0.f};
#pragma unroll
  for (int nt = 0; nt < 8; ++nt) acc[nt] = vzero;

  // ---- GEMM: 4 k-blocks x 8 n-tiles = 32 WMMAs per wave ----
#pragma unroll
  for (int kb = 0; kb < 4; ++kb) {
    v8u au = sXf[(wave * 4 + kb) * 32 + lane];          // 2x ds_load_b128
    v16bf afrag = __builtin_bit_cast(v16bf, au);
#pragma unroll
    for (int nt = 0; nt < 8; ++nt) {
      v8u bu = wsBf[(kb * 8 + nt) * 32 + lane];         // 2x global_load_b128 (L2)
      v16bf bfrag = __builtin_bit_cast(v16bf, bu);
      acc[nt] = __builtin_amdgcn_wmma_f32_16x16x32_bf16(
          false, afrag, false, bfrag, (short)0, acc[nt], false, false);
    }
  }

  // ---- Epilogue: bias + shifted softplus + dot(W2) + row reduction ----
  // C layout: lane's column n = nt*16 + ln; VGPR e holds row m = e + 8*hi.
  float bias[8], w2v[8];
#pragma unroll
  for (int nt = 0; nt < 8; ++nt) {
    bias[nt] = sB1[nt * 16 + ln];
    w2v[nt]  = sW2[nt * 16 + ln];
  }
#pragma unroll
  for (int e = 0; e < 8; ++e) {
    float s = 0.0f;
#pragma unroll
    for (int nt = 0; nt < 8; ++nt) {
      float v = acc[nt][e] + bias[nt];
      s = fmaf(ssp_fast(v), w2v[nt], s);
    }
#pragma unroll
    for (int d = 1; d < 16; d <<= 1) s += __shfl_xor(s, d, 32);
    if (ln == 0) sRow[wave][e + 8 * hi] = s;   // lanes 0 and 16 write
  }
  // same-wave LDS ordering; compiler inserts the ds waits

  // ---- Sorted-segment pooling: run-leaders issue the atomics ----
  float     rs  = 0.0f;
  long long bid = -1;
  if (lane < 16) {
    rs  = sRow[wave][ln];
    bid = batch[gRowBase + wave * 16 + ln];
  }
#pragma unroll
  for (int d = 1; d < 16; d <<= 1) {
    float     nv = __shfl_down(rs, d, 32);
    long long nb = __shfl_down(bid, d, 32);
    if ((lane + d < 16) && (nb == bid)) rs += nv;  // sorted => run-suffix sum
  }
  long long pb = __shfl_up(bid, 1, 32);
  if (lane < 16 && (lane == 0 || pb != bid)) {
    atomicAdd(out + (int)bid, rs);   // global_atomic_add_f32
  }
}

extern "C" void kernel_launch(void* const* d_in, const int* in_sizes, int n_in,
                              void* d_out, int out_size, void* d_ws, size_t ws_size,
                              hipStream_t stream) {
  (void)in_sizes; (void)n_in; (void)out_size; (void)ws_size;
  const float*     x     = (const float*)d_in[0];
  const long long* batch = (const long long*)d_in[1];   // int64 graph ids
  const float*     W1    = (const float*)d_in[2];
  const float*     b1    = (const float*)d_in[3];
  const float*     W2    = (const float*)d_in[4];
  float*           out   = (float*)d_out;
  unsigned int*    wsB   = (unsigned int*)d_ws;         // needs 32 KB

  pack_w1_kernel<<<32, 256, 0, stream>>>(W1, wsB);
  zero_out_kernel<<<(N_GRAPHS + 255) / 256, 256, 0, stream>>>(out);
  mlp_pool_kernel<<<N_NODES / ROWS_PER_BLOCK, THREADS, 0, stream>>>(
      x, batch, b1, W2, (const v8u*)wsB, out);
}
